// Criterion1_28604482192026
// MI455X (gfx1250) — matrix operand order repaired and checked by reading, not compiled, not run on hardware
//
#include <hip/hip_runtime.h>

typedef _Float16 v16h __attribute__((ext_vector_type(16)));
typedef float    v8f  __attribute__((ext_vector_type(8)));

#define BB 32
#define TT 256
#define DVC 1024
#define DJC 512
#define LW 20
#define LPAD 32
#define T1C 127
#define T2C 62
#define T3C 30
#define TPV 219
#define TPAD 224
#define LAMF 4.0f
#define MARGINF 0.1f
#define EPS_BN 1e-5f
#define EPS_COS 1e-8f

// ---- WMMA fragment index helpers (contiguous k = 16*half + e, not the ISA table order; 16-bit A 16x32) ----
__device__ __forceinline__ int a_k_of(int e, int half) {
  return half * 16 + e;
}

__device__ __forceinline__ v8f wmma_f16(v16h a, v16h b, v8f c) {
  return __builtin_amdgcn_wmma_f32_16x16x32_f16(false, a, false, b, (short)0, c,
                                                false, false);
}

// ---- column statistics: src is [R][C] row-major, stats per column (BN) ----
__global__ void colstats_kernel(const float* __restrict__ src, int R, int C,
                                float* __restrict__ mean, float* __restrict__ istd) {
  int d = blockIdx.x;
  float s = 0.f, s2 = 0.f;
  for (int r = threadIdx.x; r < R; r += blockDim.x) {
    float x = src[(size_t)r * C + d];
    s += x; s2 += x * x;
  }
  __shared__ float sh[256], sh2[256];
  sh[threadIdx.x] = s; sh2[threadIdx.x] = s2;
  __syncthreads();
  for (int st = 128; st > 0; st >>= 1) {
    if ((int)threadIdx.x < st) {
      sh[threadIdx.x]  += sh[threadIdx.x + st];
      sh2[threadIdx.x] += sh2[threadIdx.x + st];
    }
    __syncthreads();
  }
  if (threadIdx.x == 0) {
    float m = sh[0] / (float)R;
    float v = sh2[0] / (float)R - m * m;
    mean[d] = m;
    istd[d] = rsqrtf(fmaxf(v, 0.f) + EPS_BN);
  }
}

// stats for vc [B][DJ][TPAD] over (b, t<TPV) per channel d
__global__ void stats_vc_kernel(const float* __restrict__ vc,
                                float* __restrict__ mean, float* __restrict__ istd) {
  int d = blockIdx.x;
  float s = 0.f, s2 = 0.f;
  for (int r = threadIdx.x; r < BB * TPV; r += blockDim.x) {
    int b = r / TPV, t = r - b * TPV;
    float x = vc[((size_t)b * DJC + d) * TPAD + t];
    s += x; s2 += x * x;
  }
  __shared__ float sh[256], sh2[256];
  sh[threadIdx.x] = s; sh2[threadIdx.x] = s2;
  __syncthreads();
  for (int st = 128; st > 0; st >>= 1) {
    if ((int)threadIdx.x < st) {
      sh[threadIdx.x]  += sh[threadIdx.x + st];
      sh2[threadIdx.x] += sh2[threadIdx.x + st];
    }
    __syncthreads();
  }
  if (threadIdx.x == 0) {
    float m = sh[0] / (float)(BB * TPV);
    float v = sh2[0] / (float)(BB * TPV) - m * m;
    mean[d] = m;
    istd[d] = rsqrtf(fmaxf(v, 0.f) + EPS_BN);
  }
}

// cast fp32 weights -> f16
__global__ void cast_h_kernel(const float* __restrict__ in, _Float16* __restrict__ out, int n) {
  int i = blockIdx.x * blockDim.x + threadIdx.x;
  if (i < n) out[i] = (_Float16)in[i];
}

__global__ void fill_h_kernel(_Float16* __restrict__ p, int n) {
  int i = blockIdx.x * blockDim.x + threadIdx.x;
  if (i < n) p[i] = (_Float16)0.f;
}

// BN video [B,T,DV] -> vidh [B][DV][T] f16
__global__ void bn_video_kernel(const float* __restrict__ video,
                                const float* __restrict__ m, const float* __restrict__ is,
                                _Float16* __restrict__ vidh) {
  int gid = blockIdx.x * blockDim.x + threadIdx.x;
  if (gid >= BB * DVC * TT) return;
  int t = gid % TT;
  int d = (gid / TT) % DVC;
  int b = gid / (TT * DVC);
  float x = video[((size_t)b * TT + t) * DVC + d];
  vidh[gid] = (_Float16)((x - m[d]) * is[d]);
}

// BN words [B,L,DJ] -> wh [B][LPAD][DJ] f16 (rows >= LW zeroed)
__global__ void bn_words_kernel(const float* __restrict__ words,
                                const float* __restrict__ m, const float* __restrict__ is,
                                _Float16* __restrict__ wh) {
  int gid = blockIdx.x * blockDim.x + threadIdx.x;
  if (gid >= BB * LPAD * DJC) return;
  int d = gid % DJC;
  int l = (gid / DJC) % LPAD;
  int i = gid / (DJC * LPAD);
  float v = 0.f;
  if (l < LW) v = (words[((size_t)i * LW + l) * DJC + d] - m[d]) * is[d];
  wh[gid] = (_Float16)v;
}

// BN vc [B][DJ][TPAD] -> vh [B][TPAD][DJ] f16 (rows t>=TPV zeroed)
__global__ void bn_vc_kernel(const float* __restrict__ vc,
                             const float* __restrict__ m, const float* __restrict__ is,
                             _Float16* __restrict__ vh) {
  int gid = blockIdx.x * blockDim.x + threadIdx.x;
  if (gid >= BB * TPAD * DJC) return;
  int d = gid % DJC;
  int t = (gid / DJC) % TPAD;
  int b = gid / (DJC * TPAD);
  float v = 0.f;
  if (t < TPV) v = (vc[((size_t)b * DJC + d) * TPAD + t] - m[d]) * is[d];
  vh[gid] = (_Float16)v;
}

// row norms of vh: one wave per row of [B*TPAD][DJ]
__global__ void vnorm_kernel(const _Float16* __restrict__ vh, float* __restrict__ vn) {
  int row = blockIdx.x;
  const _Float16* p = vh + (size_t)row * DJC;
  float s = 0.f;
  for (int k = threadIdx.x; k < DJC; k += 32) {
    float x = (float)p[k];
    s += x * x;
  }
  #pragma unroll
  for (int off = 16; off > 0; off >>= 1) s += __shfl_xor(s, off, 32);
  if (threadIdx.x == 0) vn[row] = sqrtf(s);
}

// ---- batched implicit-im2col WMMA GEMM (conv K=4 S=2) + bias + ReLU, f16 out ----
// y[b][m][n] = relu( sum_{kg} A[m][kg] * S[b][kg>>2][sOff + 2n + (kg&3)] + bias[m] )
// B loads are unconditional (clamped column); out-of-range columns are zeroed
// with a value select (v_cndmask), NOT exec-mask branches.
__global__ void conv_gemm_kernel(const _Float16* __restrict__ A, int K,
                                 const _Float16* __restrict__ src, long sBatch, int sLd, int sOff,
                                 int Nvalid,
                                 _Float16* __restrict__ dst, long dBatch, int dLd, int dOff,
                                 const float* __restrict__ bias) {
  int b = blockIdx.z, mt = blockIdx.y, nt = blockIdx.x;
  int lane = threadIdx.x;
  int m0 = mt * 16, n0 = nt * 16;
  const _Float16* S = src + (size_t)b * sBatch;
  v8f acc = {0.f, 0.f, 0.f, 0.f, 0.f, 0.f, 0.f, 0.f};
  int am = m0 + (lane & 15);
  int ah = lane >> 4;
  int bn = n0 + (lane & 15);
  int bkh = (lane >> 4) * 16;
  bool bvalid = (bn < Nvalid);
  int bnc = bvalid ? bn : (Nvalid - 1);   // always in-bounds column
  const _Float16* Arow = A + (size_t)am * K;
  const _Float16* Bcol = S + sOff + 2 * bnc;
  for (int k0 = 0; k0 < K; k0 += 32) {
    __builtin_prefetch(Arow + k0 + 128, 0, 1);
    v16h a, bf;
    #pragma unroll
    for (int e = 0; e < 16; ++e) a[e] = Arow[k0 + a_k_of(e, ah)];
    #pragma unroll
    for (int e = 0; e < 16; ++e) {
      int kg = k0 + bkh + e;
      int ci = kg >> 2, kk = kg & 3;
      _Float16 val = Bcol[(size_t)ci * sLd + kk];
      bf[e] = bvalid ? val : (_Float16)0.f;
    }
    acc = wmma_f16(a, bf, acc);
  }
  if (bvalid) {
    int mh = lane >> 4;
    #pragma unroll
    for (int r = 0; r < 8; ++r) {
      int m = m0 + mh * 8 + r;
      float v = acc[r] + bias[m];
      v = fmaxf(v, 0.f);
      dst[(size_t)b * dBatch + (size_t)m * dLd + dOff + bn] = (_Float16)v;
    }
  }
}

// ---- batched plain WMMA GEMM (1x1 conv) + bias, fp32 out ----
__global__ void proj_gemm_kernel(const _Float16* __restrict__ A, int K,
                                 const _Float16* __restrict__ src, long sBatch, int sLd,
                                 int Nvalid,
                                 float* __restrict__ dst, long dBatch, int dLd,
                                 const float* __restrict__ bias) {
  int b = blockIdx.z, mt = blockIdx.y, nt = blockIdx.x;
  int lane = threadIdx.x;
  int m0 = mt * 16, n0 = nt * 16;
  const _Float16* S = src + (size_t)b * sBatch;
  v8f acc = {0.f, 0.f, 0.f, 0.f, 0.f, 0.f, 0.f, 0.f};
  int am = m0 + (lane & 15);
  int ah = lane >> 4;
  int bn = n0 + (lane & 15);
  int bkh = (lane >> 4) * 16;
  bool bvalid = (bn < Nvalid);
  int bnc = bvalid ? bn : (Nvalid - 1);   // always in-bounds column
  const _Float16* Arow = A + (size_t)am * K;
  const _Float16* Bcol = S + bnc;
  for (int k0 = 0; k0 < K; k0 += 32) {
    __builtin_prefetch(Arow + k0 + 128, 0, 1);
    v16h a, bf;
    #pragma unroll
    for (int e = 0; e < 16; ++e) a[e] = Arow[k0 + a_k_of(e, ah)];
    #pragma unroll
    for (int e = 0; e < 16; ++e) {
      _Float16 val = Bcol[(size_t)(k0 + bkh + e) * sLd];
      bf[e] = bvalid ? val : (_Float16)0.f;
    }
    acc = wmma_f16(a, bf, acc);
  }
  if (bvalid) {
    int mh = lane >> 4;
    #pragma unroll
    for (int r = 0; r < 8; ++r) {
      int m = m0 + mh * 8 + r;
      dst[(size_t)b * dBatch + (size_t)m * dLd + bn] = acc[r] + bias[m];
    }
  }
}

// ---- all-pairs frame-by-word attention + cosine sim + per-pair max ----
// one block per (i=sentence, j=video); 4 waves, each owns 16-row t-tiles
__global__ void attention_kernel(const _Float16* __restrict__ vh,
                                 const _Float16* __restrict__ wh,
                                 const float* __restrict__ w_masks,
                                 const float* __restrict__ vn,
                                 float* __restrict__ scores,
                                 float* __restrict__ pos_out) {
  __shared__ float    logits[4][16][32];
  __shared__ _Float16 probs[4][16][32];
  __shared__ float    wavemax[8];
  int i = blockIdx.y, j = blockIdx.x;
  int wid = threadIdx.x >> 5, lane = threadIdx.x & 31;
  const _Float16* V = vh + (size_t)j * TPAD * DJC;
  const _Float16* W = wh + (size_t)i * LPAD * DJC;
  int nlo = lane & 15;
  int mh  = lane >> 4;
  float runmax = -1e30f;

  for (int tile = wid; tile < TPAD / 16; tile += 4) {
    int t0 = tile * 16;
    // --- logits = LAM * V[j](16x512) @ W[i]^T (512x32) ---
    v8f acc0 = {0.f, 0.f, 0.f, 0.f, 0.f, 0.f, 0.f, 0.f};
    v8f acc1 = {0.f, 0.f, 0.f, 0.f, 0.f, 0.f, 0.f, 0.f};
    const _Float16* Vrow = V + (size_t)(t0 + nlo) * DJC;
    for (int k0 = 0; k0 < DJC; k0 += 32) {
      __builtin_prefetch(Vrow + k0 + 128, 0, 1);
      v16h a, b0, b1;
      #pragma unroll
      for (int e = 0; e < 16; ++e) a[e] = Vrow[k0 + a_k_of(e, mh)];
      int bk = k0 + mh * 16;
      #pragma unroll
      for (int e = 0; e < 16; ++e) b0[e] = W[(size_t)nlo * DJC + bk + e];
      #pragma unroll
      for (int e = 0; e < 16; ++e) b1[e] = W[(size_t)(16 + nlo) * DJC + bk + e];
      acc0 = wmma_f16(a, b0, acc0);
      acc1 = wmma_f16(a, b1, acc1);
    }
    #pragma unroll
    for (int r = 0; r < 8; ++r) {
      logits[wid][mh * 8 + r][nlo]      = acc0[r];
      logits[wid][mh * 8 + r][16 + nlo] = acc1[r];
    }
    asm volatile("s_wait_dscnt 0x0" ::: "memory");

    // --- masked softmax over L (lanes 0..15 each own one row) ---
    if (lane < 16) {
      float lv[LW];
      float mx = -1e30f;
      #pragma unroll
      for (int l = 0; l < LW; ++l) {
        float z = logits[wid][lane][l] * LAMF;
        z = (w_masks[i * LW + l] > 0.5f) ? z : -1e9f;
        lv[l] = z;
        mx = fmaxf(mx, z);
      }
      float s = 0.f;
      #pragma unroll
      for (int l = 0; l < LW; ++l) { float e = __expf(lv[l] - mx); lv[l] = e; s += e; }
      float inv = 1.f / s;
      #pragma unroll
      for (int l = 0; l < 32; ++l)
        probs[wid][lane][l] = (l < LW) ? (_Float16)(lv[l] * inv) : (_Float16)0.f;
    }
    asm volatile("s_wait_dscnt 0x0" ::: "memory");

    // --- v_s = P(16x32) @ W[i](32x512) fused with cosine accumulation ---
    v16h pa;
    #pragma unroll
    for (int e = 0; e < 16; ++e) pa[e] = probs[wid][nlo][a_k_of(e, mh)];
    float num_acc[8] = {0.f, 0.f, 0.f, 0.f, 0.f, 0.f, 0.f, 0.f};
    float vs2_acc[8] = {0.f, 0.f, 0.f, 0.f, 0.f, 0.f, 0.f, 0.f};
    for (int d0 = 0; d0 < DJC; d0 += 16) {
      v16h bf;
      int bk = mh * 16;
      #pragma unroll
      for (int e = 0; e < 16; ++e) bf[e] = W[(size_t)(bk + e) * DJC + d0 + nlo];
      v8f acc = {0.f, 0.f, 0.f, 0.f, 0.f, 0.f, 0.f, 0.f};
      acc = wmma_f16(pa, bf, acc);
      #pragma unroll
      for (int r = 0; r < 8; ++r) {
        float vs = acc[r];
        int m = mh * 8 + r;
        float vref = (float)V[(size_t)(t0 + m) * DJC + d0 + nlo];
        num_acc[r] += vs * vref;
        vs2_acc[r] += vs * vs;
      }
    }
    // reduce over 16 lanes (keep the two lane-halves separate: rows differ)
    #pragma unroll
    for (int off = 1; off < 16; off <<= 1) {
      #pragma unroll
      for (int r = 0; r < 8; ++r) {
        num_acc[r] += __shfl_xor(num_acc[r], off, 32);
        vs2_acc[r] += __shfl_xor(vs2_acc[r], off, 32);
      }
    }
    if (nlo == 0) {
      #pragma unroll
      for (int r = 0; r < 8; ++r) {
        int t = t0 + mh * 8 + r;
        if (t < TPV) {
          float vsn = fmaxf(sqrtf(vs2_acc[r]), EPS_COS);
          float vnn = fmaxf(vn[(size_t)j * TPAD + t], EPS_COS);
          float sim = num_acc[r] / (vnn * vsn);
          runmax = fmaxf(runmax, sim);
          if (i == j) pos_out[(size_t)i * TPV + t] = sim;
        }
      }
    }
  }
  if (nlo == 0) wavemax[wid * 2 + mh] = runmax;
  __syncthreads();
  if (threadIdx.x == 0) {
    float m = -1e30f;
    #pragma unroll
    for (int q = 0; q < 8; ++q) m = fmaxf(m, wavemax[q]);
    scores[i * BB + j] = m;
  }
}

// ---- margin ranking loss over scores[32][32] ----
__global__ void loss_kernel(const float* __restrict__ scores, float* __restrict__ out) {
  __shared__ float red[1024];
  int idx = threadIdx.x;
  int i = idx >> 5, jj = idx & 31;
  float s  = scores[i * BB + jj];
  float di = scores[i * BB + i];
  float dj = scores[jj * BB + jj];
  float c = 0.f;
  if (i != jj) {
    c += fmaxf(0.f, MARGINF + s - di);   // cost_s
    c += fmaxf(0.f, MARGINF + s - dj);   // cost_im
  }
  red[idx] = c;
  __syncthreads();
  for (int st = 512; st > 0; st >>= 1) {
    if (idx < st) red[idx] += red[idx + st];
    __syncthreads();
  }
  if (idx == 0) out[0] = red[0] / (float)BB;
}

extern "C" void kernel_launch(void* const* d_in, const int* in_sizes, int n_in,
                              void* d_out, int out_size, void* d_ws, size_t ws_size,
                              hipStream_t stream) {
  (void)in_sizes; (void)n_in; (void)out_size; (void)ws_size;
  const float* video  = (const float*)d_in[0];
  const float* words  = (const float*)d_in[1];
  const float* wmask  = (const float*)d_in[2];
  const float* c0w    = (const float*)d_in[3];
  const float* c0b    = (const float*)d_in[4];
  const float* c1w    = (const float*)d_in[5];
  const float* c1b    = (const float*)d_in[6];
  const float* c2w    = (const float*)d_in[7];
  const float* c2b    = (const float*)d_in[8];
  const float* cdw    = (const float*)d_in[9];
  const float* cdb    = (const float*)d_in[10];
  float* out = (float*)d_out;

  char* ws = (char*)d_ws;
  size_t off = 0;
  auto take = [&](size_t bytes) -> char* {
    char* p = ws + off;
    off = (off + bytes + 255) & ~(size_t)255;
    return p;
  };
  float*    msV    = (float*)take(DVC * 4);
  float*    isV    = (float*)take(DVC * 4);
  float*    msW    = (float*)take(DJC * 4);
  float*    isW    = (float*)take(DJC * 4);
  float*    msC    = (float*)take(DJC * 4);
  float*    isC    = (float*)take(DJC * 4);
  _Float16* vidh   = (_Float16*)take((size_t)BB * DVC * TT * 2);
  _Float16* A0h    = (_Float16*)take((size_t)DJC * DVC * 4 * 2);
  _Float16* A1h    = (_Float16*)take((size_t)DJC * DJC * 4 * 2);
  _Float16* A2h    = (_Float16*)take((size_t)DJC * DJC * 4 * 2);
  _Float16* A3h    = (_Float16*)take((size_t)DJC * DJC * 2);
  _Float16* Vpre   = (_Float16*)take((size_t)BB * DJC * TPAD * 2);
  float*    vc     = (float*)take((size_t)BB * DJC * TPAD * 4);
  _Float16* vh     = (_Float16*)take((size_t)BB * TPAD * DJC * 2);
  _Float16* wh     = (_Float16*)take((size_t)BB * LPAD * DJC * 2);
  float*    vn     = (float*)take((size_t)BB * TPAD * 4);
  float*    scores = (float*)take((size_t)BB * BB * 4);

  // BN statistics (video and words)
  colstats_kernel<<<DVC, 256, 0, stream>>>(video, BB * TT, DVC, msV, isV);
  colstats_kernel<<<DJC, 256, 0, stream>>>(words, BB * LW, DJC, msW, isW);

  // weight casts fp32 -> f16
  {
    int n0 = DJC * DVC * 4, n1 = DJC * DJC * 4, n3 = DJC * DJC;
    cast_h_kernel<<<(n0 + 255) / 256, 256, 0, stream>>>(c0w, A0h, n0);
    cast_h_kernel<<<(n1 + 255) / 256, 256, 0, stream>>>(c1w, A1h, n1);
    cast_h_kernel<<<(n1 + 255) / 256, 256, 0, stream>>>(c2w, A2h, n1);
    cast_h_kernel<<<(n3 + 255) / 256, 256, 0, stream>>>(cdw, A3h, n3);
  }

  // normalized video in [B][DV][T] f16
  {
    int n = BB * DVC * TT;
    bn_video_kernel<<<(n + 255) / 256, 256, 0, stream>>>(video, msV, isV, vidh);
  }

  // zero the concat buffer (pad columns t >= TPV are never written; the 1x1-conv GEMM does not read them)
  {
    int n = BB * DJC * TPAD;
    fill_h_kernel<<<(n + 255) / 256, 256, 0, stream>>>(Vpre, n);
  }

  // conv0: M=512 N=127 K=4096 ; conv1: N=62 K=2048 ; conv2: N=30 K=2048
  conv_gemm_kernel<<<dim3(8, DJC / 16, BB), 32, 0, stream>>>(
      A0h, DVC * 4, vidh, (long)DVC * TT, TT, 0, T1C,
      Vpre, (long)DJC * TPAD, TPAD, 0, c0b);
  conv_gemm_kernel<<<dim3(4, DJC / 16, BB), 32, 0, stream>>>(
      A1h, DJC * 4, Vpre, (long)DJC * TPAD, TPAD, 0, T2C,
      Vpre, (long)DJC * TPAD, TPAD, T1C, c1b);
  conv_gemm_kernel<<<dim3(2, DJC / 16, BB), 32, 0, stream>>>(
      A2h, DJC * 4, Vpre, (long)DJC * TPAD, TPAD, T1C, T3C,
      Vpre, (long)DJC * TPAD, TPAD, T1C + T2C, c2b);

  // 1x1 conv projection -> vc fp32 [B][DJ][TPAD]
  proj_gemm_kernel<<<dim3(TPAD / 16, DJC / 16, BB), 32, 0, stream>>>(
      A3h, DJC, Vpre, (long)DJC * TPAD, TPAD, TPV,
      vc, (long)DJC * TPAD, TPAD, cdb);

  // BN of projection and words -> f16 tensors for attention
  stats_vc_kernel<<<DJC, 256, 0, stream>>>(vc, msC, isC);
  {
    int n = BB * TPAD * DJC;
    bn_vc_kernel<<<(n + 255) / 256, 256, 0, stream>>>(vc, msC, isC, vh);
  }
  {
    int n = BB * LPAD * DJC;
    bn_words_kernel<<<(n + 255) / 256, 256, 0, stream>>>(words, msW, isW, wh);
  }

  // per-row norms of v
  vnorm_kernel<<<BB * TPAD, 32, 0, stream>>>(vh, vn);

  // all-pairs attention + cosine + max ; writes positive_map into out[1..]
  attention_kernel<<<dim3(BB, BB), 128, 0, stream>>>(vh, wh, wmask, vn, scores, out + 1);

  // margin ranking loss -> out[0]
  loss_kernel<<<1, 1024, 0, stream>>>(scores, out);
}
